// SelfAttention_38671885534040
// MI455X (gfx1250) — matrix operand order repaired and checked by reading, not compiled, not run on hardware
//
#include <hip/hip_runtime.h>

// ---------------------------------------------------------------------------
// Self-attention (GLM style) for MI455X / gfx1250, wave32 + WMMA bf16.
// S=2048, B=1, H=4096, NH=32, hd=128, coeff = 4.0 folded away.
// ---------------------------------------------------------------------------

#define S_LEN 2048
#define HID   4096
#define NHEAD 32
#define HD    128
#define THID  (3 * HID)   // 12288

typedef __attribute__((ext_vector_type(16))) __bf16        bf16x16;
typedef __attribute__((ext_vector_type(8)))  float         f32x8;
typedef __attribute__((ext_vector_type(4)))  float         f32x4;
typedef __attribute__((ext_vector_type(4)))  unsigned int  u32x4;

union Frag {
    bf16x16        v;
    u32x4          u[2];
    unsigned short s[16];
};

__device__ __forceinline__ unsigned short f2bf(float f) {
    unsigned int u = __builtin_bit_cast(unsigned int, f);
    u += 0x7fffu + ((u >> 16) & 1u);           // round-to-nearest-even
    return (unsigned short)(u >> 16);
}

__device__ __forceinline__ f32x8 wmma_bf16(bf16x16 a, bf16x16 b, f32x8 c) {
    return __builtin_amdgcn_wmma_f32_16x16x32_bf16(false, a, false, b,
                                                   (short)0, c, false, false);
}

// ---------------------------------------------------------------------------
// 1) fp32 -> bf16 bulk convert, 8 elems/thread (b128 in, b128 out)
// ---------------------------------------------------------------------------
__global__ void k_f32_to_bf16(const float* __restrict__ src,
                              unsigned short* __restrict__ dst, long long n) {
    long long i = (long long)blockIdx.x * blockDim.x + threadIdx.x;
    if (i * 8 >= n) return;
    f32x4 a = ((const f32x4*)src)[i * 2];
    f32x4 b = ((const f32x4*)src)[i * 2 + 1];
    u32x4 o;
    o.x = (unsigned)f2bf(a.x) | ((unsigned)f2bf(a.y) << 16);
    o.y = (unsigned)f2bf(a.z) | ((unsigned)f2bf(a.w) << 16);
    o.z = (unsigned)f2bf(b.x) | ((unsigned)f2bf(b.y) << 16);
    o.w = (unsigned)f2bf(b.z) | ((unsigned)f2bf(b.w) << 16);
    ((u32x4*)dst)[i] = o;
}

// ---------------------------------------------------------------------------
// 2) GEMM: C[M,N] = A[M,K](bf16) * W[N,K](bf16)^T + bias[N], C fp32.
//    Block 256 thr = 8 waves, block tile 128M x 128N, wave tile 64M x 32N.
//    K stepped by 32; software-pipelined with ping-pong register buffers.
// ---------------------------------------------------------------------------
__global__ void __launch_bounds__(256)
k_gemm_bf16(const unsigned short* __restrict__ A,
            const unsigned short* __restrict__ W,
            const float* __restrict__ bias,
            float* __restrict__ C, int M, int N, int K) {
    const int lane = threadIdx.x & 31;
    const int wave = threadIdx.x >> 5;
    const int hh   = lane >> 4;    // lane half (0/1)
    const int lm   = lane & 15;

    const int n0 = blockIdx.x * 128 + (wave & 3) * 32;
    const int m0 = blockIdx.y * 128 + (wave >> 2) * 64;

    const unsigned short* wrow0 = W + (size_t)(n0 + lm) * K;
    const unsigned short* wrow1 = W + (size_t)(n0 + 16 + lm) * K;
    const unsigned short* arow[4];
#pragma unroll
    for (int mt = 0; mt < 4; ++mt)
        arow[mt] = A + (size_t)(m0 + mt * 16 + lm) * K;

    f32x8 zero = {0.f, 0.f, 0.f, 0.f, 0.f, 0.f, 0.f, 0.f};
    f32x8 acc[4][2] = {{zero, zero}, {zero, zero},
                       {zero, zero}, {zero, zero}};

    Frag a0[4], b0[2], a1[4], b1[2];

    auto loadAB = [&](int k, Frag (&af)[4], Frag (&bf)[2]) {
        const unsigned short* wp0 = wrow0 + k + hh * 16;
        const unsigned short* wp1 = wrow1 + k + hh * 16;
        bf[0].u[0] = *(const u32x4*)wp0;
        bf[0].u[1] = *(const u32x4*)(wp0 + 8);
        bf[1].u[0] = *(const u32x4*)wp1;
        bf[1].u[1] = *(const u32x4*)(wp1 + 8);
#pragma unroll
        for (int mt = 0; mt < 4; ++mt) {
            const unsigned short* ap = arow[mt] + k + hh * 16;
            af[mt].u[0] = *(const u32x4*)ap;
            af[mt].u[1] = *(const u32x4*)(ap + 8);
        }
        // speculative prefetch (OOB drops silently per ISA) — no guard branch
        __builtin_prefetch(wp0 + 256, 0, 0);
        __builtin_prefetch(wp1 + 256, 0, 0);
    };

    auto compute = [&](Frag (&af)[4], Frag (&bf)[2]) {
#pragma unroll
        for (int mt = 0; mt < 4; ++mt)
#pragma unroll
            for (int nt = 0; nt < 2; ++nt)
                acc[mt][nt] = wmma_bf16(af[mt].v, bf[nt].v, acc[mt][nt]);
    };

    loadAB(0, a0, b0);
    int k = 0;
    for (; k + 64 < K; k += 64) {            // all loads in-bounds here
        loadAB(k + 32, a1, b1);
        compute(a0, b0);
        loadAB(k + 64, a0, b0);
        compute(a1, b1);
    }
    loadAB(K - 32, a1, b1);                  // k == K - 64 here
    compute(a0, b0);
    compute(a1, b1);

    const float bv0 = bias[n0 + lm];
    const float bv1 = bias[n0 + 16 + lm];
#pragma unroll
    for (int mt = 0; mt < 4; ++mt)
#pragma unroll
        for (int r = 0; r < 8; ++r) {
            int row = m0 + mt * 16 + r + 8 * hh;
            C[(size_t)row * N + n0 + lm]      = acc[mt][0][r] + bv0;
            C[(size_t)row * N + n0 + 16 + lm] = acc[mt][1][r] + bv1;
        }
}

// ---------------------------------------------------------------------------
// 3) RoPE + Q scaling. mixed fp32 [S, 12288] -> Qs/Ks bf16 [h][s][128].
//    Q pre-scaled by 1/sqrt(128) (coeff folds out of softmax).
// ---------------------------------------------------------------------------
__global__ void k_rope(const float* __restrict__ mixed,
                       const int* __restrict__ pos_ids,
                       unsigned short* __restrict__ Qs,
                       unsigned short* __restrict__ Ks) {
    int idx = blockIdx.x * blockDim.x + threadIdx.x;   // S*NH*64 threads
    int i = idx & 63;
    int h = (idx >> 6) & (NHEAD - 1);
    int s = idx >> 11;

    const float* mrow = mixed + (size_t)s * THID + h * (3 * HD);
    float q0 = mrow[i],        q1 = mrow[i + 64];
    float k0 = mrow[HD + i],   k1 = mrow[HD + i + 64];

    float pos  = (float)pos_ids[s];
    float invf = __expf(-(float)i * 0.14391156516f);   // ln(1e4)/64
    float fr   = pos * invf;
    float cs   = __cosf(fr), sn = __sinf(fr);

    // rotate_half: [i]-> -x[i+64], [i+64]-> x[i]
    float qa = q0 * cs - q1 * sn;
    float qb = q1 * cs + q0 * sn;
    float ka = k0 * cs - k1 * sn;
    float kb = k1 * cs + k0 * sn;

    const float rs = 0.08838834764831845f;             // 1/sqrt(128)
    size_t o = ((size_t)h * S_LEN + s) * HD + i;
    Qs[o]      = f2bf(qa * rs);
    Qs[o + 64] = f2bf(qb * rs);
    Ks[o]      = f2bf(ka);
    Ks[o + 64] = f2bf(kb);
}

// ---------------------------------------------------------------------------
// 4) V transpose: mixed fp32 -> Vt bf16 [h][d][s] via LDS tile.
// ---------------------------------------------------------------------------
__global__ void __launch_bounds__(256)
k_vtrans(const float* __restrict__ mixed, unsigned short* __restrict__ Vt) {
    __shared__ unsigned short lds[HD][33];
    const int h = blockIdx.y, s0 = blockIdx.x * 32, t = threadIdx.x;

    {
        int srow = t >> 3, dbase = (t & 7) * 16;
        const float* src =
            mixed + (size_t)(s0 + srow) * THID + h * (3 * HD) + 2 * HD + dbase;
#pragma unroll
        for (int e = 0; e < 16; ++e) lds[dbase + e][srow] = f2bf(src[e]);
    }
    __syncthreads();
    {
        int d = t >> 1, sb = (t & 1) * 16;
        unsigned short* dst = Vt + ((size_t)h * HD + d) * S_LEN + s0 + sb;
#pragma unroll
        for (int e = 0; e < 16; ++e) dst[e] = lds[d][sb + e];
    }
}

// ---------------------------------------------------------------------------
// 5) Flash attention. One wave per (head = bx, 16-query tile = by).
//    Scores computed TRANSPOSED (M=keys, N=queries) so the S D-fragment is
//    lane-aligned with the P A-fragment for the PV WMMA (pure per-lane pack).
//    V fragment loads issued before the softmax VALU block to hide latency.
// ---------------------------------------------------------------------------
__global__ void __launch_bounds__(32)
k_flash(const unsigned short* __restrict__ Qs,
        const unsigned short* __restrict__ Ks,
        const unsigned short* __restrict__ Vt,
        unsigned short* __restrict__ ctx) {
    const int h = blockIdx.x, qbase = blockIdx.y * 16;
    const int lane = threadIdx.x;
    const int hh = lane >> 4, lm = lane & 15;
    const int query = qbase + lm;

    // Q as B-fragments (K=d chunk of 32, N=16 queries)
    Frag qf[4];
    {
        const unsigned short* qrow = Qs + ((size_t)h * S_LEN + query) * HD;
#pragma unroll
        for (int c = 0; c < 4; ++c) {
            const unsigned short* p = qrow + c * 32 + hh * 8;
            qf[c].u[0] = *(const u32x4*)p;
            qf[c].u[1] = *(const u32x4*)(p + 16);
        }
    }

    f32x8 zero = {0.f, 0.f, 0.f, 0.f, 0.f, 0.f, 0.f, 0.f};
    f32x8 o[8] = {zero, zero, zero, zero, zero, zero, zero, zero};
    float m = -1e30f, lsum = 0.f;

    const unsigned short* kbaseptr = Ks + (size_t)h * S_LEN * HD;
    const unsigned short* vbaseptr = Vt + (size_t)h * HD * S_LEN;

    const int jend = (qbase + 15) >> 5;           // key blocks of 32
    for (int j = 0; j <= jend; ++j) {
        const int kb0 = j * 32;

        // S^T = Ktile(16k x 32d per wmma) * Q^T, two key halves
        f32x8 s0 = zero, s1 = zero;
#pragma unroll
        for (int c = 0; c < 4; ++c) {
            Frag ka, kb;
            const unsigned short* kr0 =
                kbaseptr + (size_t)(kb0 + lm) * HD + c * 32 + hh * 8;
            ka.u[0] = *(const u32x4*)kr0;
            ka.u[1] = *(const u32x4*)(kr0 + 16);
            const unsigned short* kr1 =
                kbaseptr + (size_t)(kb0 + 16 + lm) * HD + c * 32 + hh * 8;
            kb.u[0] = *(const u32x4*)kr1;
            kb.u[1] = *(const u32x4*)(kr1 + 16);
            s0 = wmma_bf16(ka.v, qf[c].v, s0);
            s1 = wmma_bf16(kb.v, qf[c].v, s1);
        }

        // V fragments for this key block: independent of softmax -> issue now
        Frag vf[8];
#pragma unroll
        for (int c = 0; c < 8; ++c) {
            const unsigned short* vp =
                vbaseptr + (size_t)(c * 16 + lm) * S_LEN + kb0 + hh * 8;
            vf[c].u[0] = *(const u32x4*)vp;
            vf[c].u[1] = *(const u32x4*)(vp + 16);
        }

        // causal mask + row max (per lane: query = lm; keys in vgprs)
        float tmax = -1e30f;
#pragma unroll
        for (int r = 0; r < 8; ++r) {
            int key0 = kb0 + r + 8 * hh;
            if (key0 > query)      s0[r] = -1e30f;
            if (key0 + 16 > query) s1[r] = -1e30f;
            tmax = fmaxf(tmax, fmaxf(s0[r], s1[r]));
        }
        tmax = fmaxf(tmax, __shfl_xor(tmax, 16, 32));
        float mnew = fmaxf(m, tmax);
        float corr = __expf(m - mnew);

        float rsum = 0.f;
#pragma unroll
        for (int r = 0; r < 8; ++r) {
            float p0 = __expf(s0[r] - mnew);
            float p1 = __expf(s1[r] - mnew);
            s0[r] = p0; s1[r] = p1;
            rsum += p0 + p1;
        }
        rsum += __shfl_xor(rsum, 16, 32);
        lsum = lsum * corr + rsum;
        m = mnew;

        // rescale O: O vgpr r <-> query row r+8*hh; corr lives in lane qrow
#pragma unroll
        for (int r = 0; r < 8; ++r) {
            float sc = __shfl(corr, r + 8 * hh, 32);
#pragma unroll
            for (int c = 0; c < 8; ++c) o[c][r] *= sc;
        }

        // pack P as A-fragment (pure per-lane: layouts line up)
        Frag pa;
#pragma unroll
        for (int i = 0; i < 8; ++i) {
            pa.s[i]     = f2bf(s0[i]);
            pa.s[8 + i] = f2bf(s1[i]);
        }

        // O += P * Vtile
#pragma unroll
        for (int c = 0; c < 8; ++c)
            o[c] = wmma_bf16(pa.v, vf[c].v, o[c]);
    }

    // normalize + store context bf16 at [s][h*128 + d]
#pragma unroll
    for (int r = 0; r < 8; ++r) {
        float lq  = __shfl(lsum, r + 8 * hh, 32);
        float inv = 1.0f / lq;
        int row = qbase + r + 8 * hh;
        unsigned short* dst = ctx + (size_t)row * HID + h * HD + lm;
#pragma unroll
        for (int c = 0; c < 8; ++c) dst[c * 16] = f2bf(o[c][r] * inv);
    }
}

// ---------------------------------------------------------------------------
// Launch
// ---------------------------------------------------------------------------
extern "C" void kernel_launch(void* const* d_in, const int* in_sizes, int n_in,
                              void* d_out, int out_size, void* d_ws,
                              size_t ws_size, hipStream_t stream) {
    const float* hidden  = (const float*)d_in[0];
    const int*   pos_ids = (const int*)d_in[1];
    // d_in[2]: mask (bool) — causal mask computed analytically, unused
    const float* qkv_w   = (const float*)d_in[3];
    const float* qkv_b   = (const float*)d_in[4];
    const float* dense_w = (const float*)d_in[5];
    const float* dense_b = (const float*)d_in[6];
    float*       out     = (float*)d_out;

    char* base = (char*)d_ws;
    size_t off = 0;
    auto take = [&](size_t bytes) -> void* {
        void* p = base + off;
        off += (bytes + 255) & ~(size_t)255;
        return p;
    };

    unsigned short* hbf   = (unsigned short*)take((size_t)S_LEN * HID * 2);
    unsigned short* wqkv  = (unsigned short*)take((size_t)THID * HID * 2);
    unsigned short* wdns  = (unsigned short*)take((size_t)HID * HID * 2);
    float*          mixed = (float*)take((size_t)S_LEN * THID * 4);
    unsigned short* Qs    = (unsigned short*)take((size_t)NHEAD * S_LEN * HD * 2);
    unsigned short* Ks    = (unsigned short*)take((size_t)NHEAD * S_LEN * HD * 2);
    unsigned short* Vt    = (unsigned short*)take((size_t)NHEAD * S_LEN * HD * 2);
    unsigned short* ctx   = (unsigned short*)take((size_t)S_LEN * HID * 2);

    // 1) convert to bf16 (8 elems / thread)
    {
        long long n1 = (long long)S_LEN * HID;
        long long n2 = (long long)THID * HID;
        long long n3 = (long long)HID * HID;
        k_f32_to_bf16<<<(unsigned)((n1 / 8 + 255) / 256), 256, 0, stream>>>(hidden, hbf, n1);
        k_f32_to_bf16<<<(unsigned)((n2 / 8 + 255) / 256), 256, 0, stream>>>(qkv_w, wqkv, n2);
        k_f32_to_bf16<<<(unsigned)((n3 / 8 + 255) / 256), 256, 0, stream>>>(dense_w, wdns, n3);
    }

    // 2) QKV projection: mixed = hidden * qkv_w^T + qkv_b   (fp32 out)
    k_gemm_bf16<<<dim3(THID / 128, S_LEN / 128), 256, 0, stream>>>(
        hbf, wqkv, qkv_b, mixed, S_LEN, THID, HID);

    // 3) RoPE (+ fold 1/(sqrt(hd)*coeff) * coeff into Q)
    k_rope<<<(S_LEN * NHEAD * 64) / 256, 256, 0, stream>>>(mixed, pos_ids, Qs, Ks);

    // 4) V transpose to [h][d][s]
    k_vtrans<<<dim3(S_LEN / 32, NHEAD), 256, 0, stream>>>(mixed, Vt);

    // 5) flash attention -> context bf16 [s][h*hd+d]
    k_flash<<<dim3(NHEAD, S_LEN / 16), 32, 0, stream>>>(Qs, Ks, Vt, ctx);

    // 6) output projection: out = ctx * dense_w^T + dense_b  (fp32 out)
    k_gemm_bf16<<<dim3(HID / 128, S_LEN / 128), 256, 0, stream>>>(
        ctx, wdns, dense_b, out, S_LEN, HID, HID);
}
